// Bilstm_70626442215456
// MI455X (gfx1250) — matrix operand order repaired and checked by reading, not compiled, not run on hardware
//
#include <hip/hip_runtime.h>
#include <hip/hip_bf16.h>

// Problem constants (match reference)
#define TT 512
#define BB 64
#define EE 512
#define HDD 512
#define GG 2048          // 4*HD
#define NCLS 16
#define START_TAG 14
#define STOP_TAG 15
#define NEGV (-10000.0f)

typedef __bf16 bf16t;
typedef __attribute__((ext_vector_type(16))) __bf16 v16bf;
typedef __attribute__((ext_vector_type(8)))  __bf16 v8bf;
typedef __attribute__((ext_vector_type(8)))  float  v8f;

// ---------------------------------------------------------------------------
// WMMA helpers (CDNA5 16x16x32 bf16 -> f32)
// ---------------------------------------------------------------------------
__device__ __forceinline__ v8f wmma_bf16(v16bf a, v16bf b, v8f c) {
  // 8 args: (neg_a, A, neg_b, B, c_mod, C, reuse_a, reuse_b)
  return __builtin_amdgcn_wmma_f32_16x16x32_bf16(false, a, false, b, (short)0, c,
                                                 false, false);
}

// A-matrix 16x32 bf16, contiguous k order (matches B):
//   lane 0-15 : row=lane,    elems 0..15 = K 0..15
//   lane16-31 : row=lane-16, elems 0..15 = K 16..31
// => per lane one contiguous 32-byte run at (half*16), read as two 16B halves.
__device__ __forceinline__ v16bf pack_a(const bf16t* p) {
  v8bf lo = *reinterpret_cast<const v8bf*>(p);
  v8bf hi = *reinterpret_cast<const v8bf*>(p + 8);
  v16bf r;
#pragma unroll
  for (int i = 0; i < 8; ++i) { r[i] = lo[i]; r[i + 8] = hi[i]; }
  return r;
}

// B-matrix 32x16 bf16: lane 0-15 col=lane K=0..15, lane 16-31 col=lane-16
// K=16..31 -> one contiguous 32-byte run per lane (row-major [N, K] weights).
// 16-byte-granular version (used for LDS where only 16B alignment is sure).
__device__ __forceinline__ v16bf pack_b16(const bf16t* p) {
  v8bf lo = *reinterpret_cast<const v8bf*>(p);
  v8bf hi = *reinterpret_cast<const v8bf*>(p + 8);
  v16bf r;
#pragma unroll
  for (int i = 0; i < 8; ++i) { r[i] = lo[i]; r[i + 8] = hi[i]; }
  return r;
}

__device__ __forceinline__ float sigm(float x) { return 1.0f / (1.0f + __expf(-x)); }
// tanh via exp: saturates to +-1 for |x| large, no NaN.
__device__ __forceinline__ float fast_tanh(float x) {
  return 1.0f - 2.0f / (__expf(2.0f * x) + 1.0f);
}

// LDS byte-offset of a generic pointer that refers to shared memory.
__device__ __forceinline__ unsigned lds_offset_of(const void* p) {
  return (unsigned)(unsigned long long)
      (__attribute__((address_space(3))) const void*)p;
}

// ---------------------------------------------------------------------------
// LSTM step kernel: fused fwd(t) + bwd(T-1-t), one layer at a time.
// Grid: 64 blocks (0..31 fwd, 32..63 bwd), block = 128 threads (4 waves).
// Block owns 16 hidden units (jt), wave owns 16 batch rows.
// W_hh strip (64 rows x 512 bf16 = 64KB) is staged into LDS with gfx1250
// async copies (ASYNCcnt) and shared by all 4 waves.
// ---------------------------------------------------------------------------
struct StepDir {
  const bf16t* a1; int k1;      // x segment 1 at this timestep, [64, k1], ld=lda
  const bf16t* a2; int k2;      // x segment 2 (layer1 second half), or null
  int lda;
  const bf16t* w_ih; int ldw;   // [2048, Din] row-major, this direction
  const bf16t* h_in;            // [64, 512] bf16
  const bf16t* w_hh;            // [2048, 512] row-major, this direction
  const float* bias;            // [2048] = b_ih + b_hh, this direction
  float*       c;               // [64, 512] f32, updated in place
  bf16t*       h_out;           // [64, 512] bf16 (ping-pong partner of h_in)
  bf16t*       hs_out;          // [64, 512] slice of layer-output sequence
};

// GEMM over one K segment, weights streamed from global (L2-resident).
__device__ __forceinline__ void gemm4(const bf16t* A, int lda, int klen,
                                      const bf16t* W, int ldw,
                                      int jt, int m0, int lane,
                                      v8f& g0, v8f& g1, v8f& g2, v8f& g3) {
  const int l15 = lane & 15, half = lane >> 4;
  const bf16t* ar = A + (size_t)(m0 + l15) * lda + half * 16;
  const bf16t* w0 = W + (size_t)(0 * HDD + jt * 16 + l15) * ldw + half * 16;
  const bf16t* w1 = W + (size_t)(1 * HDD + jt * 16 + l15) * ldw + half * 16;
  const bf16t* w2 = W + (size_t)(2 * HDD + jt * 16 + l15) * ldw + half * 16;
  const bf16t* w3 = W + (size_t)(3 * HDD + jt * 16 + l15) * ldw + half * 16;
  for (int k0 = 0; k0 < klen; k0 += 32) {
    // keep weights warm near the WGP; speculative, safe if past the end
    __builtin_prefetch(w0 + k0 + 32, 0, 3);
    v16bf a = pack_a(ar + k0);
    g0 = wmma_bf16(a, *reinterpret_cast<const v16bf*>(w0 + k0), g0);
    g1 = wmma_bf16(a, *reinterpret_cast<const v16bf*>(w1 + k0), g1);
    g2 = wmma_bf16(a, *reinterpret_cast<const v16bf*>(w2 + k0), g2);
    g3 = wmma_bf16(a, *reinterpret_cast<const v16bf*>(w3 + k0), g3);
  }
}

// GEMM over the recurrent K segment, weights read from LDS strip
// (row r = gate*16 + n, 512 bf16 per row).
__device__ __forceinline__ void gemm4_lds(const bf16t* A, int lda,
                                          const bf16t* Wl,
                                          int m0, int lane,
                                          v8f& g0, v8f& g1, v8f& g2, v8f& g3) {
  const int l15 = lane & 15, half = lane >> 4;
  const bf16t* ar = A + (size_t)(m0 + l15) * lda + half * 16;
  const bf16t* w0 = Wl + (size_t)(0 * 16 + l15) * HDD + half * 16;
  const bf16t* w1 = Wl + (size_t)(1 * 16 + l15) * HDD + half * 16;
  const bf16t* w2 = Wl + (size_t)(2 * 16 + l15) * HDD + half * 16;
  const bf16t* w3 = Wl + (size_t)(3 * 16 + l15) * HDD + half * 16;
  for (int k0 = 0; k0 < HDD; k0 += 32) {
    v16bf a = pack_a(ar + k0);
    g0 = wmma_bf16(a, pack_b16(w0 + k0), g0);
    g1 = wmma_bf16(a, pack_b16(w1 + k0), g1);
    g2 = wmma_bf16(a, pack_b16(w2 + k0), g2);
    g3 = wmma_bf16(a, pack_b16(w3 + k0), g3);
  }
}

__global__ __launch_bounds__(128) void lstm_step_kernel(StepDir F, StepDir R) {
  extern __shared__ bf16t sh_whh[];              // 64 rows x 512 bf16 = 64KB
  StepDir P = (blockIdx.x < 32) ? F : R;
  const int jt   = blockIdx.x & 31;
  const int wave = threadIdx.x >> 5;
  const int lane = threadIdx.x & 31;
  const int m0   = wave * 16;

  // ---- Stage this block's W_hh strip into LDS with async copies ----------
  {
    const unsigned lds_base = lds_offset_of(sh_whh);
    const int tid = threadIdx.x;                 // 0..127
#pragma unroll 4
    for (int i = 0; i < 32; ++i) {               // 4096 16B chunks / 128 thr
      const int chunk = tid + i * 128;
      const int row   = chunk >> 6;              // 64 x 16B chunks per row
      const int colb  = (chunk & 63) * 16;       // byte offset in row
      const int g     = row >> 4, n = row & 15;
      const bf16t* src = P.w_hh +
          (size_t)(g * HDD + jt * 16 + n) * HDD + (colb >> 1);
      const unsigned dst = lds_base + (unsigned)(row * (HDD * 2) + colb);
      asm volatile("global_load_async_to_lds_b128 %0, %1, off"
                   :: "v"(dst), "v"(src) : "memory");
    }
    asm volatile("s_wait_asynccnt 0x0" ::: "memory");
    __syncthreads();
  }

  v8f g0 = {0.f,0.f,0.f,0.f,0.f,0.f,0.f,0.f};
  v8f g1 = g0, g2 = g0, g3 = g0;

  gemm4(P.a1, P.lda, P.k1, P.w_ih, P.ldw, jt, m0, lane, g0, g1, g2, g3);
  if (P.k2 > 0)
    gemm4(P.a2, P.lda, P.k2, P.w_ih + P.k1, P.ldw, jt, m0, lane, g0, g1, g2, g3);
  gemm4_lds(P.h_in, HDD, sh_whh, m0, lane, g0, g1, g2, g3);

  // C/D layout: elem r of v8f -> row m0 + r + 8*half, col = jt*16 + (lane&15)
  const int n    = jt * 16 + (lane & 15);
  const int half = lane >> 4;
#pragma unroll
  for (int r = 0; r < 8; ++r) {
    const int m   = m0 + r + half * 8;
    const int off = m * HDD + n;
    float iv = sigm(g0[r] + P.bias[n]);
    float fv = sigm(g1[r] + P.bias[HDD + n]);
    float gv = fast_tanh(g2[r] + P.bias[2 * HDD + n]);
    float ov = sigm(g3[r] + P.bias[3 * HDD + n]);
    float cn = fv * P.c[off] + iv * gv;
    float hn = ov * fast_tanh(cn);
    P.c[off] = cn;
    bf16t hb = (bf16t)hn;
    P.h_out[off]  = hb;
    P.hs_out[off] = hb;
  }
}

// ---------------------------------------------------------------------------
// Final linear: feats[b][t][c] = concat(H1f,H1b)[t*64+b, :] @ linW.T + lin_b
// M = T*B = 32768 rows, N = 16 (one tile), K = 1024 (two 512 segments).
// Grid 512 blocks x 128 threads; wave owns one 16-row M tile.
// ---------------------------------------------------------------------------
__global__ __launch_bounds__(128) void linear_kernel(const bf16t* Hf, const bf16t* Hb,
                                                     const bf16t* W, const float* bias,
                                                     float* feats) {
  const int mt   = blockIdx.x * 4 + (threadIdx.x >> 5);
  const int lane = threadIdx.x & 31;
  const int l15  = lane & 15, half = lane >> 4;
  const int m0   = mt * 16;
  v8f acc = {0.f,0.f,0.f,0.f,0.f,0.f,0.f,0.f};
  {
    const bf16t* ar = Hf + (size_t)(m0 + l15) * HDD + half * 16;
    const bf16t* wr = W + (size_t)l15 * (2 * HDD) + half * 16;
    for (int k0 = 0; k0 < HDD; k0 += 32)
      acc = wmma_bf16(pack_a(ar + k0), *reinterpret_cast<const v16bf*>(wr + k0), acc);
  }
  {
    const bf16t* ar = Hb + (size_t)(m0 + l15) * HDD + half * 16;
    const bf16t* wr = W + (size_t)l15 * (2 * HDD) + HDD + half * 16;
    for (int k0 = 0; k0 < HDD; k0 += 32)
      acc = wmma_bf16(pack_a(ar + k0), *reinterpret_cast<const v16bf*>(wr + k0), acc);
  }
#pragma unroll
  for (int r = 0; r < 8; ++r) {
    const int m = m0 + r + half * 8;
    const int b = m & (BB - 1);
    const int t = m >> 6;
    feats[(size_t)b * TT * NCLS + (size_t)t * NCLS + l15] = acc[r] + bias[l15];
  }
}

// ---------------------------------------------------------------------------
// Viterbi: one wave per batch item, lanes = tags (lanes 16-31 shadow 0-15).
// ---------------------------------------------------------------------------
__global__ __launch_bounds__(32) void viterbi_kernel(const float* feats, const float* trans,
                                                     int* bp, float* out) {
  const int b    = blockIdx.x;
  const int lane = threadIdx.x;
  const int tag  = lane & 15;
  float tr[NCLS];
#pragma unroll
  for (int p = 0; p < NCLS; ++p) tr[p] = trans[tag * NCLS + p];  // trans[next][prev]
  float fv = (tag == START_TAG) ? 0.0f : NEGV;
  const float* fb  = feats + (size_t)b * TT * NCLS;
  int*         bpb = bp    + (size_t)b * TT * NCLS;
  for (int t = 0; t < TT; ++t) {
    float best = -3.4e38f; int bi = 0;
#pragma unroll
    for (int p = 0; p < NCLS; ++p) {
      float s = __shfl(fv, p, 32) + tr[p];
      if (s > best) { best = s; bi = p; }     // strict > => first-max (jnp.argmax)
    }
    float nv = best + fb[t * NCLS + tag];
    if (lane < NCLS) bpb[t * NCLS + tag] = bi;
    fv = nv;
  }
  float term = fv + trans[STOP_TAG * NCLS + tag];
  float best = -3.4e38f; int bi = 0;
#pragma unroll
  for (int p = 0; p < NCLS; ++p) {
    float s = __shfl(term, p, 32);
    if (s > best) { best = s; bi = p; }
  }
  if (lane == 0) {
    out[b] = best;
    float* path = out + BB + (size_t)b * TT;
    int tg = bi;
    for (int t = TT - 1; t >= 0; --t) { path[t] = (float)tg; tg = bpb[t * NCLS + tg]; }
  }
}

// ---------------------------------------------------------------------------
// Small utility kernels
// ---------------------------------------------------------------------------
__global__ void embed_kernel(const int* sent, const float* emb, bf16t* X0, long total) {
  for (long i = (long)blockIdx.x * blockDim.x + threadIdx.x; i < total;
       i += (long)gridDim.x * blockDim.x) {
    int  e  = (int)(i & (EE - 1));
    long tb = i >> 9;                 // EE == 512
    int  b  = (int)(tb & (BB - 1));
    int  t  = (int)(tb >> 6);         // BB == 64
    int  tok = sent[b * TT + t];
    X0[i] = (bf16t)emb[(long)tok * EE + e];
  }
}

__global__ void conv_kernel(const float* src, bf16t* dst, long n) {
  for (long i = (long)blockIdx.x * blockDim.x + threadIdx.x; i < n;
       i += (long)gridDim.x * blockDim.x)
    dst[i] = (bf16t)src[i];
}

__global__ void copy_kernel(const float* src, float* dst, long n) {
  for (long i = (long)blockIdx.x * blockDim.x + threadIdx.x; i < n;
       i += (long)gridDim.x * blockDim.x)
    dst[i] = src[i];
}

__global__ void add_kernel(const float* a, const float* b, float* dst, long n) {
  for (long i = (long)blockIdx.x * blockDim.x + threadIdx.x; i < n;
       i += (long)gridDim.x * blockDim.x)
    dst[i] = a[i] + b[i];
}

// ---------------------------------------------------------------------------
// Host orchestration (graph-capture safe: stream-only, deterministic)
// ---------------------------------------------------------------------------
static inline unsigned nblk(long n) {
  long b = (n + 255) / 256;
  if (b > 4096) b = 4096;
  return (unsigned)b;
}

extern "C" void kernel_launch(void* const* d_in, const int* in_sizes, int n_in,
                              void* d_out, int out_size, void* d_ws, size_t ws_size,
                              hipStream_t stream) {
  (void)in_sizes; (void)n_in; (void)out_size; (void)ws_size;
  const int*   sentence  = (const int*)  d_in[0];
  const float* embedding = (const float*)d_in[1];
  const float* w_ih_l0   = (const float*)d_in[2];
  const float* w_hh_l0   = (const float*)d_in[3];
  const float* b_ih_l0   = (const float*)d_in[4];
  const float* b_hh_l0   = (const float*)d_in[5];
  const float* w_ih_l1   = (const float*)d_in[6];
  const float* w_hh_l1   = (const float*)d_in[7];
  const float* b_ih_l1   = (const float*)d_in[8];
  const float* b_hh_l1   = (const float*)d_in[9];
  const float* lin_w     = (const float*)d_in[10];
  const float* lin_b     = (const float*)d_in[11];
  const float* trans     = (const float*)d_in[12];
  const float* h0        = (const float*)d_in[13];
  const float* c0        = (const float*)d_in[14];

  // Bump allocator over d_ws, 256B aligned (keeps all WMMA loads 32B aligned).
  char* base = (char*)d_ws;
  size_t off = 0;
  auto alloc = [&](size_t bytes) -> void* {
    void* p = base + off;
    off = (off + bytes + 255) & ~(size_t)255;
    return p;
  };

  bf16t* X0   = (bf16t*)alloc((size_t)TT * BB * EE  * 2);
  bf16t* H0f  = (bf16t*)alloc((size_t)TT * BB * HDD * 2);
  bf16t* H0b  = (bf16t*)alloc((size_t)TT * BB * HDD * 2);
  bf16t* H1f  = (bf16t*)alloc((size_t)TT * BB * HDD * 2);
  bf16t* H1b  = (bf16t*)alloc((size_t)TT * BB * HDD * 2);
  bf16t* wih0 = (bf16t*)alloc((size_t)2 * GG * EE        * 2);
  bf16t* whh0 = (bf16t*)alloc((size_t)2 * GG * HDD       * 2);
  bf16t* wih1 = (bf16t*)alloc((size_t)2 * GG * (2 * HDD) * 2);
  bf16t* whh1 = (bf16t*)alloc((size_t)2 * GG * HDD       * 2);
  bf16t* linw = (bf16t*)alloc((size_t)NCLS * (2 * HDD)   * 2);
  float* bias0 = (float*)alloc((size_t)2 * GG * 4);
  float* bias1 = (float*)alloc((size_t)2 * GG * 4);
  bf16t* hfA = (bf16t*)alloc((size_t)BB * HDD * 2);
  bf16t* hfB = (bf16t*)alloc((size_t)BB * HDD * 2);
  bf16t* hbA = (bf16t*)alloc((size_t)BB * HDD * 2);
  bf16t* hbB = (bf16t*)alloc((size_t)BB * HDD * 2);
  float* cf  = (float*)alloc((size_t)BB * HDD * 4);
  float* cb  = (float*)alloc((size_t)BB * HDD * 4);
  float* feats = (float*)alloc((size_t)BB * TT * NCLS * 4);
  int*   bp    = (int*)  alloc((size_t)BB * TT * NCLS * 4);

  // Prep: embed + convert weights + combine biases
  long nx = (long)TT * BB * EE;
  embed_kernel<<<nblk(nx), 256, 0, stream>>>(sentence, embedding, X0, nx);
  conv_kernel<<<nblk(2L*GG*EE),      256, 0, stream>>>(w_ih_l0, wih0, 2L*GG*EE);
  conv_kernel<<<nblk(2L*GG*HDD),     256, 0, stream>>>(w_hh_l0, whh0, 2L*GG*HDD);
  conv_kernel<<<nblk(2L*GG*2*HDD),   256, 0, stream>>>(w_ih_l1, wih1, 2L*GG*2*HDD);
  conv_kernel<<<nblk(2L*GG*HDD),     256, 0, stream>>>(w_hh_l1, whh1, 2L*GG*HDD);
  conv_kernel<<<nblk((long)NCLS*2*HDD), 256, 0, stream>>>(lin_w, linw, (long)NCLS*2*HDD);
  add_kernel<<<nblk(2L*GG), 256, 0, stream>>>(b_ih_l0, b_hh_l0, bias0, 2L*GG);
  add_kernel<<<nblk(2L*GG), 256, 0, stream>>>(b_ih_l1, b_hh_l1, bias1, 2L*GG);

  for (int layer = 0; layer < 2; ++layer) {
    const bf16t* wih  = layer ? wih1  : wih0;
    const bf16t* whh  = layer ? whh1  : whh0;
    const float* bias = layer ? bias1 : bias0;
    const int    din  = layer ? 2 * HDD : EE;
    bf16t* HsF = layer ? H1f : H0f;
    bf16t* HsB = layer ? H1b : H0b;
    const size_t BH = (size_t)BB * HDD;
    const int sf = layer * 2, sb = layer * 2 + 1;
    conv_kernel<<<nblk((long)BH), 256, 0, stream>>>(h0 + sf * BH, hfA, (long)BH);
    conv_kernel<<<nblk((long)BH), 256, 0, stream>>>(h0 + sb * BH, hbA, (long)BH);
    copy_kernel<<<nblk((long)BH), 256, 0, stream>>>(c0 + sf * BH, cf,  (long)BH);
    copy_kernel<<<nblk((long)BH), 256, 0, stream>>>(c0 + sb * BH, cb,  (long)BH);

    bf16t *hfi = hfA, *hfo = hfB, *hbi = hbA, *hbo = hbB;
    for (int t = 0; t < TT; ++t) {
      const int tb = TT - 1 - t;
      StepDir F, R;
      if (layer == 0) {
        F.a1 = X0 + (size_t)t  * BB * EE; F.k1 = EE; F.a2 = nullptr; F.k2 = 0; F.lda = EE;
        R.a1 = X0 + (size_t)tb * BB * EE; R.k1 = EE; R.a2 = nullptr; R.k2 = 0; R.lda = EE;
      } else {
        F.a1 = H0f + (size_t)t  * BH; F.a2 = H0b + (size_t)t  * BH;
        R.a1 = H0f + (size_t)tb * BH; R.a2 = H0b + (size_t)tb * BH;
        F.k1 = HDD; F.k2 = HDD; F.lda = HDD;
        R.k1 = HDD; R.k2 = HDD; R.lda = HDD;
      }
      F.w_ih = wih;                        F.ldw = din;
      R.w_ih = wih + (size_t)GG * din;     R.ldw = din;
      F.w_hh = whh;
      R.w_hh = whh + (size_t)GG * HDD;
      F.bias = bias;       R.bias = bias + GG;
      F.h_in = hfi; F.h_out = hfo; F.c = cf; F.hs_out = HsF + (size_t)t  * BH;
      R.h_in = hbi; R.h_out = hbo; R.c = cb; R.hs_out = HsB + (size_t)tb * BH;
      lstm_step_kernel<<<64, 128, 64 * 1024, stream>>>(F, R);
      bf16t* tmp = hfi; hfi = hfo; hfo = tmp;
      tmp = hbi; hbi = hbo; hbo = tmp;
    }
  }

  linear_kernel<<<(TT * BB / 16) / 4, 128, 0, stream>>>(H1f, H1b, linw, lin_b, feats);
  viterbi_kernel<<<BB, 32, 0, stream>>>(feats, trans, bp, (float*)d_out);
}
